// BsPINN_69904887710540
// MI455X (gfx1250) — compile-verified, matrix operand order checked by reading, not run on hardware
//
#include <hip/hip_runtime.h>
#include <hip/hip_bf16.h>

// ---------------------------------------------------------------------------
// BsPINN fused forward for MI455X (gfx1250, wave32, WMMA f16->f32).
//
// Net: X[N,2] -> 1024 (dense) -> 1024 (dense) -> 1024 (blockdiag 2x512)
//      -> 1024 (blockdiag 4x256) -> 1, tanh between, fp32 in/out.
//
// Strategy: weights converted once per launch to f16, transposed (Wt[n][k])
// and compacted (block-diag layers store only nonzero blocks) into d_ws.
// One fused kernel keeps a 64-row activation tile resident in LDS (two
// 64x1032-half buffers, 264 KB) and runs every layer with
// v_wmma_f32_16x16x32_f16. Activations never touch HBM.
// tanh uses the gfx1250 hardware TRANS op (v_tanh_f32) when available.
// ---------------------------------------------------------------------------

typedef _Float16 half8  __attribute__((ext_vector_type(8)));
typedef _Float16 v16h   __attribute__((ext_vector_type(16)));
typedef float    v8f    __attribute__((ext_vector_type(8)));

#define HDIM     1024
#define NROWS    64            // rows per workgroup
#define STRIDEH  1032          // LDS row stride in halves: 1024 + 8 pad
                               // -> 2064 B/row = 516 dwords, 516 % 64 == 4:
                               // b128 A-frag reads hit the 2-cycle LDS minimum.

// ---- hardware tanh (TRANS unit, co-executes with XDL WMMA) -----------------
__device__ __forceinline__ float fast_tanh(float x) {
#if __has_builtin(__builtin_amdgcn_tanhf)
    return __builtin_amdgcn_tanhf(x);       // v_tanh_f32
#else
    return tanhf(x);
#endif
}

// ---- fragment load: 16 f16 as two 8-runs 16 halves apart = two b128 loads (lo at +0, hi at +32B)
__device__ __forceinline__ v16h load_frag(const _Float16* p) {
    const half8* q = (const half8*)p;
    half8 lo = q[0];
    half8 hi = q[2];
    return __builtin_shufflevector(lo, hi, 0,1,2,3,4,5,6,7,8,9,10,11,12,13,14,15);
}

__device__ __forceinline__ v8f splat8(float x) {
    v8f r;
#pragma unroll
    for (int i = 0; i < 8; ++i) r[i] = x;
    return r;
}

__device__ __forceinline__ v8f wmma_f16(v16h a, v16h b, v8f c) {
    return __builtin_amdgcn_wmma_f32_16x16x32_f16(
        /*neg_a=*/false, a, /*neg_b=*/false, b,
        /*c_mod=*/(short)0, c, /*reuse_a=*/false, /*reuse_b=*/false);
}

// ---- one hidden layer: out = tanh(in @ W + b), W given transposed/compacted.
// KT      : k-tiles (K/32) per output column (32 dense, 16 L2, 8 L3)
// NTPB    : n-tiles (16 cols each) per diagonal block (64 = dense)
// WST     : row stride of Wt in halves (= KT*32)
template<int KT, int NTPB, int WST>
__device__ __forceinline__ void layer_wmma(
    const _Float16* __restrict__ inb,   // LDS, 64 x STRIDEH halves
    _Float16* __restrict__ outb,        // LDS, 64 x STRIDEH halves
    const _Float16* __restrict__ Wt,    // global (L2-resident), [1024][WST]
    const float* __restrict__ bias,     // global fp32 [1024]
    int wave, int lane)
{
    const int mt    = wave >> 1;        // 0..3  : 16-row M tile
    const int nbase = (wave & 1) * 32;  // 0|32  : 32 of 64 N tiles
    const int l16   = lane & 15;
    const int hi8   = (lane >> 4) << 3; // K sub-offset per half-wave
    const int rhi   = (lane >> 4) << 3; // M sub-offset for C layout stores

    const _Float16* aRow = inb + (mt * 16 + l16) * STRIDEH;

    for (int g = 0; g < 8; ++g) {
        const int n0   = nbase + g * 4;               // first of 4 n-tiles
        const int kOff = (n0 / NTPB) * (KT * 32);     // block-diag K base

        v8f acc[4];
#pragma unroll
        for (int t = 0; t < 4; ++t)
            acc[t] = splat8(bias[(n0 + t) * 16 + l16]);

        const _Float16* bRow[4];
#pragma unroll
        for (int t = 0; t < 4; ++t)
            bRow[t] = Wt + ((size_t)((n0 + t) * 16 + l16)) * WST;

#pragma unroll 4
        for (int k = 0; k < KT; ++k) {
            const int ko = k * 32 + hi8;
            v16h a = load_frag(aRow + kOff + ko);     // ds_read_b128 x2
#pragma unroll
            for (int t = 0; t < 4; ++t) {
                v16h b = load_frag(bRow[t] + ko);     // global_load_b128 x2
                acc[t] = wmma_f16(a, b, acc[t]);      // v_wmma_f32_16x16x32_f16
            }
        }

        // C layout: VGPR v -> row mt*16 + rhi + v, col (n)*16 + l16
#pragma unroll
        for (int t = 0; t < 4; ++t) {
            const int col = (n0 + t) * 16 + l16;
            _Float16* orow = outb + (mt * 16 + rhi) * STRIDEH + col;
#pragma unroll
            for (int v = 0; v < 8; ++v)
                orow[v * STRIDEH] = (_Float16)fast_tanh(acc[t][v]);
        }
    }
}

// ---------------------------------------------------------------------------
__global__ __launch_bounds__(256)
void bspinn_fused(const float* __restrict__ X,
                  const float* __restrict__ W0, const float* __restrict__ b0,
                  const _Float16* __restrict__ W1t, const float* __restrict__ b1,
                  const _Float16* __restrict__ W2t, const float* __restrict__ b2,
                  const _Float16* __restrict__ W3t, const float* __restrict__ b3,
                  const float* __restrict__ Wl, const float* __restrict__ bl,
                  float* __restrict__ out)
{
    extern __shared__ _Float16 smem[];
    _Float16* buf0 = smem;
    _Float16* buf1 = smem + NROWS * STRIDEH;

    const int t    = threadIdx.x;
    const int wave = t >> 5;
    const int lane = t & 31;
    const int row0 = blockIdx.x * NROWS;

    // ---- layer 0: h0 = tanh(norm(X) @ W0 + b0), K=2 -> plain VALU ----------
    // norm: xn0 = x0/pi - 1, xn1 = 2*x1 - 1
#pragma unroll 4
    for (int it = 0; it < (NROWS * HDIM) / 256; ++it) {
        const int flat = it * 256 + t;
        const int r = flat >> 10;
        const int j = flat & (HDIM - 1);
        const float x0 = X[(size_t)(row0 + r) * 2 + 0];
        const float x1 = X[(size_t)(row0 + r) * 2 + 1];
        const float xn0 = x0 * 0.31830988618379067f - 1.0f;
        const float xn1 = 2.0f * x1 - 1.0f;
        const float v = fast_tanh(xn0 * W0[j] + xn1 * W0[HDIM + j] + b0[j]);
        buf0[r * STRIDEH + j] = (_Float16)v;
    }
    __syncthreads();

    // ---- layer 1: dense 1024 -> 1024 ---------------------------------------
    layer_wmma<32, 64, 1024>(buf0, buf1, W1t, b1, wave, lane);
    __syncthreads();

    // ---- layer 2: block-diagonal, 2 blocks of 512x512 ----------------------
    layer_wmma<16, 32, 512>(buf1, buf0, W2t, b2, wave, lane);
    __syncthreads();

    // ---- layer 3: block-diagonal, 4 blocks of 256x256 ----------------------
    layer_wmma<8, 16, 256>(buf0, buf1, W3t, b3, wave, lane);
    __syncthreads();

    // ---- output layer: out[r] = h3[r,:] . Wl + bl --------------------------
    float* part = (float*)buf0;             // buf0 free now
    if (t < NROWS) part[t] = bl[0];
    __syncthreads();

    {
        const int r = t & 63;               // lanes of a wave span 32 rows
        const int q = t >> 6;               // quarter of the 1024 columns
        const _Float16* hrow = buf1 + r * STRIDEH + q * 256;
        const float* wq = Wl + q * 256;
        float s = 0.0f;
#pragma unroll 8
        for (int i = 0; i < 256; ++i)
            s += (float)hrow[i] * wq[i];
        atomicAdd(&part[r], s);             // ds_add_f32
    }
    __syncthreads();

    if (t < NROWS) out[row0 + t] = part[t];
}

// ---------------------------------------------------------------------------
// Weight prep: fp32 -> f16, transpose to Wt[n][k], compact block-diag layers.
#define W1T_ELEMS (1024 * 1024)
#define W2T_ELEMS (1024 * 512)
#define W3T_ELEMS (1024 * 256)
#define PREP_TOTAL (W1T_ELEMS + W2T_ELEMS + W3T_ELEMS)

__global__ __launch_bounds__(256)
void bspinn_prep(const float* __restrict__ W1,
                 const float* __restrict__ W2,
                 const float* __restrict__ W3,
                 _Float16* __restrict__ w1t,
                 _Float16* __restrict__ w2t,
                 _Float16* __restrict__ w3t)
{
    const int i = blockIdx.x * blockDim.x + threadIdx.x;
    if (i < W1T_ELEMS) {
        const int j = i >> 10, k = i & 1023;            // w1t[j][k] = W1[k][j]
        w1t[i] = (_Float16)W1[(size_t)k * 1024 + j];
    } else if (i < W1T_ELEMS + W2T_ELEMS) {
        const int ii = i - W1T_ELEMS;
        const int j = ii >> 9, kk = ii & 511;           // block = j/512
        const int blk = j >> 9;
        w2t[ii] = (_Float16)W2[(size_t)(blk * 512 + kk) * 1024 + j];
    } else if (i < PREP_TOTAL) {
        const int ii = i - (W1T_ELEMS + W2T_ELEMS);
        const int j = ii >> 8, kk = ii & 255;           // block = j/256
        const int blk = j >> 8;
        w3t[ii] = (_Float16)W3[(size_t)(blk * 256 + kk) * 1024 + j];
    }
}

// ---------------------------------------------------------------------------
extern "C" void kernel_launch(void* const* d_in, const int* in_sizes, int n_in,
                              void* d_out, int out_size, void* d_ws, size_t ws_size,
                              hipStream_t stream) {
    const float* X  = (const float*)d_in[0];
    const float* W0 = (const float*)d_in[1];
    const float* b0 = (const float*)d_in[2];
    const float* W1 = (const float*)d_in[3];
    const float* b1 = (const float*)d_in[4];
    const float* W2 = (const float*)d_in[5];
    const float* b2 = (const float*)d_in[6];
    const float* W3 = (const float*)d_in[7];
    const float* b3 = (const float*)d_in[8];
    const float* Wl = (const float*)d_in[9];
    const float* bl = (const float*)d_in[10];
    float* out = (float*)d_out;

    _Float16* w1t = (_Float16*)d_ws;
    _Float16* w2t = w1t + W1T_ELEMS;
    _Float16* w3t = w2t + W2T_ELEMS;

    const int N = in_sizes[0] / 2;          // 65536

    // f16 weight prep (weights are L2-resident afterwards: ~3.5 MB)
    {
        const int blocks = (PREP_TOTAL + 255) / 256;
        bspinn_prep<<<blocks, 256, 0, stream>>>(W1, W2, W3, w1t, w2t, w3t);
    }

    // fused MLP: 64 rows / workgroup, 264 KB dynamic LDS (<= 320 KB/WGP)
    const size_t ldsBytes = 2u * NROWS * STRIDEH * sizeof(_Float16);
    static_assert(2u * NROWS * STRIDEH * sizeof(_Float16) <= 320u * 1024u, "LDS");
    hipFuncSetAttribute((const void*)bspinn_fused,
                        hipFuncAttributeMaxDynamicSharedMemorySize,
                        (int)ldsBytes);

    bspinn_fused<<<N / NROWS, 256, ldsBytes, stream>>>(
        X, W0, b0, w1t, b1, w2t, b2, w3t, b3, Wl, bl, out);
}
